// Attention_57131654972090
// MI455X (gfx1250) — matrix operand order repaired and checked by reading, not compiled, not run on hardware
//
#include <hip/hip_runtime.h>
#include <hip/hip_bf16.h>

#define B_    64
#define T_    1024
#define ENC_  512
#define QDIM_ 1024
#define ADIM_ 128
#define NFILT_ 32
#define KSIZE_ 31

typedef __attribute__((ext_vector_type(16))) __bf16 v16bf;
typedef __attribute__((ext_vector_type(8)))  float  v8f;

// round-to-nearest-even f32 -> bf16, packed pair into one dword
__device__ __forceinline__ unsigned pack2bf16(float lo, float hi) {
    union { float f; unsigned u; } a, b;
    a.f = lo; b.f = hi;
    unsigned ra = a.u + 0x7FFFu + ((a.u >> 16) & 1u);
    unsigned rb = b.u + 0x7FFFu + ((b.u >> 16) & 1u);
    return (ra >> 16) | (rb & 0xFFFF0000u);
}

// tanh via hardware v_exp_f32 + v_rcp_f32 (no precise-div fixup sequence)
__device__ __forceinline__ float fast_tanh(float x) {
    x = fminf(15.0f, fmaxf(-15.0f, x));
    float e = __expf(2.0f * x);
    return (e - 1.0f) * __builtin_amdgcn_rcpf(e + 1.0f);
}

// ---------------------------------------------------------------------------
// Kernel 1: qloc[b,j] = tanh(hidden[b]@Wq + bq)[j] + tanh((aw[b]@conv_w[15])@Wloc)[j]
// (SAME-pad conv over spatial width 1 keeps only the center tap k=15)
// ---------------------------------------------------------------------------
__global__ void __launch_bounds__(ADIM_)
prep_kernel(const float* __restrict__ hidden, const float* __restrict__ aw,
            const float* __restrict__ Wq, const float* __restrict__ bq,
            const float* __restrict__ conv_w, const float* __restrict__ Wloc,
            float* __restrict__ qloc) {
    __shared__ float convs[NFILT_];
    const int b = blockIdx.x;
    const int j = threadIdx.x;                 // 0..127
    if (j < NFILT_) {
        const float* cw  = conv_w + (KSIZE_ / 2) * T_ * NFILT_ + j;
        const float* awb = aw + b * T_;
        float s = 0.f;
        for (int t = 0; t < T_; ++t) s += awb[t] * cw[t * NFILT_];
        convs[j] = s;
    }
    __syncthreads();
    const float* hb = hidden + b * QDIM_;
    float s = bq[j];
    for (int k = 0; k < QDIM_; ++k) s += hb[k] * Wq[k * ADIM_ + j];
    float pq = fast_tanh(s);
    float cl = 0.f;
    #pragma unroll
    for (int f = 0; f < NFILT_; ++f) cl += convs[f] * Wloc[f * ADIM_ + j];
    qloc[b * ADIM_ + j] = pq + fast_tanh(cl);
}

// ---------------------------------------------------------------------------
// Kernel 2: pre-swizzle Wm (512x128 f32) into bf16 WMMA B-fragments.
// Fragment f = kt*8+nt. Lanes 0-15: N=lane, K=kt*32+{0..15} (two per dword);
// lanes 16-31: N=lane-16, K=kt*32+{16..31}.
// ---------------------------------------------------------------------------
__global__ void __launch_bounds__(32)
pack_wm_kernel(const float* __restrict__ Wm, unsigned* __restrict__ wsB) {
    const int f    = blockIdx.x;               // 0..127
    const int lane = threadIdx.x;              // 0..31
    const int kt = f >> 3, nt = f & 7;
    const int half = lane >> 4, lr = lane & 15;
    const int n = nt * 16 + lr;
    unsigned* dst = wsB + f * 256 + lane * 8;
    #pragma unroll
    for (int d = 0; d < 8; ++d) {
        int k = kt * 32 + half * 16 + (d << 1);
        dst[d] = pack2bf16(Wm[k * ADIM_ + n], Wm[(k + 1) * ADIM_ + n]);
    }
}

// A-fragment builder: half 0 holds K {0..15}, half 1 {16..31} (matches packed B)
struct AFrag { union { unsigned u[8]; v16bf v; }; };
__device__ __forceinline__ AFrag make_afrag(const float* __restrict__ arow, int k0h) {
    const float4* p0 = (const float4*)(arow + k0h);
    const float4* p1 = (const float4*)(arow + k0h + 8);
    float4 f0 = p0[0], f1 = p0[1], f2 = p1[0], f3 = p1[1];
    AFrag a;
    a.u[0] = pack2bf16(f0.x, f0.y);  a.u[1] = pack2bf16(f0.z, f0.w);
    a.u[2] = pack2bf16(f1.x, f1.y);  a.u[3] = pack2bf16(f1.z, f1.w);
    a.u[4] = pack2bf16(f2.x, f2.y);  a.u[5] = pack2bf16(f2.z, f2.w);
    a.u[6] = pack2bf16(f3.x, f3.y);  a.u[7] = pack2bf16(f3.z, f3.w);
    return a;
}

// ---------------------------------------------------------------------------
// Kernel 3: fused  pm = tanh(mem@Wm+bm);  e = sum_n tanh(qloc+pm)*Vw + Vb
// 256 threads = 8 waves; each wave owns 32 rows x 128 cols (two A fragments
// sharing every B fragment -> 2 WMMAs per ds-read pair). K swept in 16 steps
// of 32 with double-buffered 8KB LDS staging of the pre-swizzled bf16 Wm.
// ---------------------------------------------------------------------------
__global__ void __launch_bounds__(256)
gemm_energy_kernel(const float* __restrict__ memory, const unsigned* __restrict__ wsB,
                   const float* __restrict__ qloc, const float* __restrict__ bm,
                   const float* __restrict__ Vw, const float* __restrict__ Vb,
                   float* __restrict__ energies) {
    __shared__ __align__(16) unsigned ldsB[2][2048];   // double buffer, 8KB each
    __shared__ float s_q[ADIM_], s_bm[ADIM_], s_vw[ADIM_];

    const int tid  = threadIdx.x;
    const int wave = tid >> 5, lane = tid & 31;
    const int half = lane >> 4, lr = lane & 15;
    const int rowBlock = blockIdx.x * 256;             // 256 rows per block, single batch
    const int b = rowBlock / T_;

    if (tid < ADIM_) {
        s_q[tid]  = qloc[b * ADIM_ + tid];
        s_bm[tid] = bm[tid];
        s_vw[tid] = Vw[tid];
    }

    const int rowBase = rowBlock + wave * 32;
    const float* arow0 = memory + (size_t)(rowBase + lr) * ENC_;        // rows 0..15
    const float* arow1 = memory + (size_t)(rowBase + 16 + lr) * ENC_;   // rows 16..31

    v8f acc0[8] = {}, acc1[8] = {};

    // prologue: stage kt=0
    {
        const uint4* src = (const uint4*)(wsB);
        uint4*       dst = (uint4*)ldsB[0];
        dst[tid] = src[tid]; dst[tid + 256] = src[tid + 256];
    }
    __syncthreads();

    for (int kt = 0; kt < 16; ++kt) {
        const int cur = kt & 1;
        // stage next K-step into the other buffer while this one computes
        if (kt + 1 < 16) {
            const uint4* src = (const uint4*)(wsB + (kt + 1) * 2048);
            uint4*       dst = (uint4*)ldsB[cur ^ 1];
            dst[tid] = src[tid]; dst[tid + 256] = src[tid + 256];
        }

        const int k0h = kt * 32 + half * 16;
        AFrag a0 = make_afrag(arow0, k0h);
        AFrag a1 = make_afrag(arow1, k0h);
        __builtin_prefetch(arow0 + kt * 32 + 128, 0, 0);
        __builtin_prefetch(arow1 + kt * 32 + 128, 0, 0);

        // preload all 8 B fragments, then run the WMMA chain (amortize ds waits)
        uint4 bl[16];
        #pragma unroll
        for (int nt = 0; nt < 8; ++nt) {
            const uint4* bp = (const uint4*)(ldsB[cur] + nt * 256 + lane * 8);
            bl[2 * nt]     = bp[0];
            bl[2 * nt + 1] = bp[1];
        }
        #pragma unroll
        for (int nt = 0; nt < 8; ++nt) {
            union { uint4 q[2]; v16bf v; } bfr;
            bfr.q[0] = bl[2 * nt]; bfr.q[1] = bl[2 * nt + 1];
            acc0[nt] = __builtin_amdgcn_wmma_f32_16x16x32_bf16(
                false, a0.v, false, bfr.v, (short)0, acc0[nt], false, false);
            acc1[nt] = __builtin_amdgcn_wmma_f32_16x16x32_bf16(
                false, a1.v, false, bfr.v, (short)0, acc1[nt], false, false);
        }
        __syncthreads();
    }

    // Epilogue: D layout => VGPR r, lanes 0-15: (M=r, N=lane); lanes 16-31: (M=r+8, N=lane-16)
    const float vb = Vb[0];
    #pragma unroll
    for (int m = 0; m < 2; ++m) {
        #pragma unroll
        for (int r = 0; r < 8; ++r) {
            float partial = 0.f;
            #pragma unroll
            for (int nt = 0; nt < 8; ++nt) {
                int col = nt * 16 + lr;
                float v  = m ? acc1[nt][r] : acc0[nt][r];
                float pm = fast_tanh(v + s_bm[col]);
                partial += fast_tanh(s_q[col] + pm) * s_vw[col];
            }
            partial += __shfl_xor(partial, 1, 32);
            partial += __shfl_xor(partial, 2, 32);
            partial += __shfl_xor(partial, 4, 32);
            partial += __shfl_xor(partial, 8, 32);
            if (lr == 0) energies[rowBase + m * 16 + r + half * 8] = partial + vb;
        }
    }
}

// ---------------------------------------------------------------------------
// Kernel 4: softmax over T per batch + context = w * memory  (bandwidth bound)
// ---------------------------------------------------------------------------
__global__ void __launch_bounds__(256)
softmax_ctx_kernel(const float* __restrict__ memory, const float* __restrict__ energies,
                   float* __restrict__ out_ctx, float* __restrict__ out_w) {
    __shared__ float s_w[T_];
    __shared__ float s_red[8];
    const int b = blockIdx.x, tid = threadIdx.x;
    const float* eb = energies + b * T_;

    float ev[4], mx = -1e30f;
    #pragma unroll
    for (int i = 0; i < 4; ++i) { ev[i] = eb[tid + i * 256]; mx = fmaxf(mx, ev[i]); }
    for (int off = 16; off; off >>= 1) mx = fmaxf(mx, __shfl_xor(mx, off, 32));
    if ((tid & 31) == 0) s_red[tid >> 5] = mx;
    __syncthreads();
    mx = s_red[0];
    #pragma unroll
    for (int w = 1; w < 8; ++w) mx = fmaxf(mx, s_red[w]);

    float sum = 0.f;
    #pragma unroll
    for (int i = 0; i < 4; ++i) {
        float e = __expf(ev[i] - mx);
        s_w[tid + i * 256] = e;
        sum += e;
    }
    for (int off = 16; off; off >>= 1) sum += __shfl_xor(sum, off, 32);
    __syncthreads();                       // s_red reads above & s_w writes done
    if ((tid & 31) == 0) s_red[tid >> 5] = sum;
    __syncthreads();
    sum = 0.f;
    #pragma unroll
    for (int w = 0; w < 8; ++w) sum += s_red[w];
    const float inv = 1.0f / sum;

    #pragma unroll
    for (int i = 0; i < 4; ++i) out_w[b * T_ + tid + i * 256] = s_w[tid + i * 256] * inv;

    const float4* mem4 = (const float4*)(memory  + (size_t)b * T_ * ENC_);
    float4*       ctx4 = (float4*)      (out_ctx + (size_t)b * T_ * ENC_);
    for (int idx = tid; idx < T_ * (ENC_ / 4); idx += 256) {
        int t = idx >> 7;                  // 128 float4 per row
        float w = s_w[t] * inv;
        float4 m = mem4[idx];
        float4 o; o.x = m.x * w; o.y = m.y * w; o.z = m.z * w; o.w = m.w * w;
        ctx4[idx] = o;
    }
}

// ---------------------------------------------------------------------------
extern "C" void kernel_launch(void* const* d_in, const int* in_sizes, int n_in,
                              void* d_out, int out_size, void* d_ws, size_t ws_size,
                              hipStream_t stream) {
    const float* hidden = (const float*)d_in[0];
    const float* memory = (const float*)d_in[1];
    const float* awcat  = (const float*)d_in[2];
    const float* Wq     = (const float*)d_in[3];
    const float* bq     = (const float*)d_in[4];
    const float* Wm     = (const float*)d_in[5];
    const float* bm     = (const float*)d_in[6];
    const float* conv_w = (const float*)d_in[7];
    const float* Wloc   = (const float*)d_in[8];
    const float* Vw     = (const float*)d_in[9];
    const float* Vb     = (const float*)d_in[10];

    float* out_ctx = (float*)d_out;                          // (B,T,ENC)
    float* out_w   = out_ctx + (size_t)B_ * T_ * ENC_;       // (B,T)

    float*    ws_q = (float*)d_ws;                           // B*ADIM
    float*    ws_e = ws_q + B_ * ADIM_;                      // B*T
    unsigned* ws_B = (unsigned*)(ws_e + B_ * T_);            // 32768 dwords (bf16 Wm, swizzled)

    prep_kernel<<<B_, ADIM_, 0, stream>>>(hidden, awcat, Wq, bq, conv_w, Wloc, ws_q);
    pack_wm_kernel<<<128, 32, 0, stream>>>(Wm, ws_B);
    gemm_energy_kernel<<<(B_ * T_) / 256, 256, 0, stream>>>(memory, ws_B, ws_q, bm, Vw, Vb, ws_e);
    softmax_ctx_kernel<<<B_, 256, 0, stream>>>(memory, ws_e, out_ctx, out_w);
}
